// ParticleNet_35055523070556
// MI455X (gfx1250) — matrix operand order repaired and checked by reading, not compiled, not run on hardware
//
#include <hip/hip_runtime.h>
#include <hip/hip_bf16.h>

typedef __attribute__((ext_vector_type(16))) _Float16 v16h;
typedef __attribute__((ext_vector_type(8)))  float    v8f;

#define BN_EPS 1e-5f
#define BIGF   1000000000.0f
#define BATCH  256
#define NPTS   128
#define KNN    16

// ---------------------------------------------------------------- utilities
__global__ void zero_f32(float* __restrict__ p, int n) {
  int i = blockIdx.x * blockDim.x + threadIdx.x;
  if (i < n) p[i] = 0.f;
}

// -------------------------------------------- mask, counts, repack to NHWC
// features (B,7,N), points (B,2,N) -> mask(B*N), counts(B), fts0(B*N,32 zero-pad), pts0(B*N,2)
__global__ void mask_kernel(const float* __restrict__ feats,
                            const float* __restrict__ pts,
                            float* __restrict__ mask,
                            float* __restrict__ counts,
                            float* __restrict__ fts0,
                            float* __restrict__ pts0) {
  int i = blockIdx.x * blockDim.x + threadIdx.x; // b*N + n
  if (i >= BATCH * NPTS) return;
  int b = i / NPTS, n = i - b * NPTS;
  float f[7], s = 0.f;
  for (int c = 0; c < 7; ++c) { f[c] = feats[(b * 7 + c) * NPTS + n]; s += fabsf(f[c]); }
  float m = (s != 0.f) ? 1.f : 0.f;
  mask[i] = m;
  atomicAdd(&counts[b], m);
  for (int c = 0; c < 7; ++c)  fts0[i * 32 + c] = f[c] * m;
  for (int c = 7; c < 32; ++c) fts0[i * 32 + c] = 0.f;          // K-pad for WMMA
  for (int c = 0; c < 2; ++c)  pts0[i * 2 + c] = pts[(b * 2 + c) * NPTS + n] * m;
}

// -------------------------------------------- BN training-mode statistics
__global__ void bn_stats(const float* __restrict__ X, int ld, long long M, int C,
                         float* __restrict__ sums) {
  __shared__ float lds[1024]; // 2*C <= 1024 (C max 384)
  for (int c = threadIdx.x; c < 2 * C; c += blockDim.x) lds[c] = 0.f;
  __syncthreads();
  long long total  = M * (long long)C;
  long long stride = (long long)gridDim.x * blockDim.x;
  for (long long t = (long long)blockIdx.x * blockDim.x + threadIdx.x; t < total; t += stride) {
    long long row = t / C; int c = (int)(t - row * C);
    float x = X[row * ld + c];
    atomicAdd(&lds[c], x);
    atomicAdd(&lds[C + c], x * x);
  }
  __syncthreads();
  for (int c = threadIdx.x; c < 2 * C; c += blockDim.x)
    if (lds[c] != 0.f) atomicAdd(&sums[c], lds[c]);
}

__global__ void bn_finalize(const float* __restrict__ sums,
                            const float* __restrict__ g, const float* __restrict__ b,
                            int C, float invM, float* __restrict__ sb) {
  int c = blockIdx.x * blockDim.x + threadIdx.x;
  if (c >= C) return;
  float mean = sums[c] * invM;
  float var  = sums[C + c] * invM - mean * mean;
  float sc   = g[c] * rsqrtf(var + BN_EPS);
  sb[c]     = sc;
  sb[C + c] = b[c] - mean * sc;
}

__global__ void bn_apply(float* __restrict__ X, int ld, long long M, int C,
                         const float* __restrict__ sb, int relu,
                         const float* __restrict__ mask) {
  long long t = (long long)blockIdx.x * blockDim.x + threadIdx.x;
  if (t >= M * (long long)C) return;
  long long row = t / C; int c = (int)(t - row * C);
  float y = X[row * ld + c] * sb[c] + sb[C + c];
  if (relu) y = fmaxf(y, 0.f);
  if (mask) y *= mask[row];
  X[row * ld + c] = y;
}

// ------------------------------------------------------------------- kNN
__global__ void knn_kernel(const float* __restrict__ coords, int ld, int D,
                           const float* __restrict__ mask, int* __restrict__ idx) {
  extern __shared__ float sh[]; // NPTS * D
  int b = blockIdx.x;
  const float* cb = coords + (long long)b * NPTS * ld;
  const float* mb = mask + b * NPTS;
  for (int t = threadIdx.x; t < NPTS * D; t += blockDim.x) {
    int m = t / D, d = t - m * D;
    sh[t] = cb[m * ld + d] + (1.f - mb[m]) * BIGF;
  }
  __syncthreads();
  int n = threadIdx.x;
  if (n >= NPTS) return;
  float bd[KNN]; int bi[KNN];
  for (int j = 0; j < KNN; ++j) { bd[j] = 3.0e38f; bi[j] = 0; }
  for (int m = 0; m < NPTS; ++m) {
    if (m == n) continue;
    float dist = 0.f;
    for (int d = 0; d < D; ++d) { float df = sh[n * D + d] - sh[m * D + d]; dist += df * df; }
    if (dist < bd[KNN - 1]) {
      int j = KNN - 1;
      while (j > 0 && bd[j - 1] > dist) { bd[j] = bd[j - 1]; bi[j] = bi[j - 1]; --j; }
      bd[j] = dist; bi[j] = m;
    }
  }
  for (int j = 0; j < KNN; ++j) idx[((long long)b * NPTS + n) * KNN + j] = bi[j];
}

// --------------------------------------------------- edge feature assembly
// X[(b,n,j), 0..C) = ctr ; X[., C..2C) = nbr-ctr ; X[., 2C..ldx) = 0 (K pad)
__global__ void edge_feat(const float* __restrict__ fts, int ld, int C, int ldx,
                          const int* __restrict__ idx, float* __restrict__ X) {
  long long rows = (long long)BATCH * NPTS * KNN;
  long long r = (long long)blockIdx.x * blockDim.x + threadIdx.x;
  if (r >= rows) return;
  long long bn = r / KNN;
  long long b  = bn / NPTS;
  int nb = idx[r];
  const float* ctr = fts + bn * ld;
  const float* nbr = fts + (b * NPTS + nb) * ld;
  float* out = X + r * ldx;
  for (int c = 0; c < C; ++c) {
    float cc = ctr[c];
    out[c]     = cc;
    out[C + c] = nbr[c] - cc;
  }
  for (int c = 2 * C; c < ldx; ++c) out[c] = 0.f;
}

// ------------------------------------------ weight pack: f32 -> f16 B-frags
// Fragment layout, this code's contiguous k order (16-bit B 32x16): lane = half*16+n,
// lane's 16 halves = K pairs {half*16 + 2v, +1}. Zero-pad K->Kp, N->Ncp.
__global__ void pack_w(const float* __restrict__ W, int K, int Nc,
                       int nK /*Kp/32*/, int nTiles, _Float16* __restrict__ out) {
  int t = blockIdx.x * blockDim.x + threadIdx.x;
  if (t >= nTiles * nK * 32) return;
  int lane = t & 31;
  int tmp  = t >> 5;
  int kc   = tmp % nK;
  int nt   = tmp / nK;
  int half = lane >> 4, r = lane & 15;
  int col  = nt * 16 + r;
  v16h frag;
#pragma unroll
  for (int v = 0; v < 8; ++v) {
    int kb = kc * 32 + half * 16 + 2 * v;
    frag[2 * v]     = (col < Nc && kb     < K) ? (_Float16)W[(long long)col * K + kb]     : (_Float16)0.f;
    frag[2 * v + 1] = (col < Nc && kb + 1 < K) ? (_Float16)W[(long long)col * K + kb + 1] : (_Float16)0.f;
  }
  ((v16h*)out)[t] = frag;
}

// ------------------------------------------------------------ WMMA GEMM
// C[M,Ncp] = A[M,Kp](lda=Kp) * Bp^T (+bias)(+relu).
// Block = one 32-row M tile x up to 8 N tiles (one per wave).
// Each wave: two WMMAs per K-step sharing one packed-B fragment.
// A tiles double-buffered in LDS (fragment order), coop f32->f16 staging
// software-pipelined so the next A pair + B fragment are in flight across
// the single per-iteration barrier.
// Requires: M % 32 == 0, Kp % 32 == 0 (guaranteed by padding at producers).
__global__ void __launch_bounds__(256)
wmma_gemm_kernel(const float* __restrict__ A, int lda,
                 const _Float16* __restrict__ Bp,
                 const float* __restrict__ bias,
                 float* __restrict__ C, int ldc,
                 int Kp, int nTiles, int relu) {
  __shared__ alignas(32) _Float16 atile[2][2][512]; // [buf][mhalf][lane*16]
  int tid  = threadIdx.x;
  int wave = tid >> 5, lane = tid & 31;
  int mt    = blockIdx.x;                       // 32-row tile index
  int ntRaw = blockIdx.y * 8 + wave;
  int nt    = (ntRaw < nTiles) ? ntRaw : (nTiles - 1); // clamp; dup waves don't store
  int half = lane >> 4, r = lane & 15;

  // cooperative A staging: thread t -> (dest lane dl, fragment pair vv)
  int dl = tid & 31, vv = tid >> 5;
  int arow = mt * 32 + (dl & 15);
  int aka  = 2 * vv + (dl >> 4) * 16;
  const float* aptr0 = A + (long long)arow * lda + aka;
  const float* aptr1 = aptr0 + 16LL * lda;

  int nK = Kp >> 5;
  const v16h* bptr = (const v16h*)Bp + (long long)nt * nK * 32 + lane;

  float2 a0 = *(const float2*)aptr0; aptr0 += 32;   // prologue prefetch
  float2 a1 = *(const float2*)aptr1; aptr1 += 32;
  v8f acc0 = {}, acc1 = {};
  for (int kc = 0; kc < nK; ++kc) {
    int cur = kc & 1;
    _Float16* d0 = &atile[cur][0][dl * 16 + vv * 2];
    _Float16* d1 = &atile[cur][1][dl * 16 + vv * 2];
    d0[0] = (_Float16)a0.x; d0[1] = (_Float16)a0.y;
    d1[0] = (_Float16)a1.x; d1[1] = (_Float16)a1.y;
    if (kc + 1 < nK) {                          // uniform branch: prefetch next
      a0 = *(const float2*)aptr0; aptr0 += 32;
      a1 = *(const float2*)aptr1; aptr1 += 32;
    }
    v16h bv = bptr[(long long)kc * 32];         // B load in flight across barrier
    __syncthreads();
    v16h av0 = *(const v16h*)&atile[cur][0][lane * 16];  // ds_load_b128
    v16h av1 = *(const v16h*)&atile[cur][1][lane * 16];
    acc0 = __builtin_amdgcn_wmma_f32_16x16x32_f16(false, av0, false, bv,
                                                  (short)0, acc0, false, false);
    acc1 = __builtin_amdgcn_wmma_f32_16x16x32_f16(false, av1, false, bv,
                                                  (short)0, acc1, false, false);
  }
  if (ntRaw < nTiles) {
    int col = nt * 16 + r;
#pragma unroll
    for (int v = 0; v < 8; ++v) {               // D: VGPR v -> M = v + 8*half
      int orow = mt * 32 + v + 8 * half;
      float y0 = acc0[v];
      float y1 = acc1[v];
      if (bias) { y0 += bias[col]; y1 += bias[col]; }
      if (relu) { y0 = fmaxf(y0, 0.f); y1 = fmaxf(y1, 0.f); }
      C[(long long)orow * ldc + col] = y0;
      C[(long long)(orow + 16) * ldc + col] = y1;
    }
  }
}

// ---- combine: BN+ReLU of Y3, mean over k, + BN(shortcut), * mask ----
__global__ void combine_kernel(const float* __restrict__ Y3, int C,
                               const float* __restrict__ sb3,
                               const float* __restrict__ SC,
                               const float* __restrict__ sb_sc,
                               const float* __restrict__ mask,
                               float* __restrict__ fused, int coff) {
  long long rowsBN = (long long)BATCH * NPTS;
  long long t = (long long)blockIdx.x * blockDim.x + threadIdx.x;
  if (t >= rowsBN * C) return;
  long long row = t / C; int c = (int)(t - row * C);
  float sc3 = sb3[c], bi3 = sb3[C + c];
  const float* yp = Y3 + row * (long long)KNN * C + c;
  float s = 0.f;
  for (int j = 0; j < KNN; ++j) s += fmaxf(yp[(long long)j * C] * sc3 + bi3, 0.f);
  s *= (1.f / KNN);
  float scv = SC[row * (long long)C + c] * sb_sc[c] + sb_sc[C + c];
  fused[row * 448 + coff + c] = (s + scv) * mask[row];
}

// ---- pool: BN+ReLU on fusion output, * mask, sum over N / counts ----
__global__ void pool_kernel(const float* __restrict__ Yf, const float* __restrict__ sb,
                            const float* __restrict__ mask, const float* __restrict__ counts,
                            float* __restrict__ pooled, int C) {
  int b = blockIdx.x;
  float cnt = fmaxf(counts[b], 1.f);
  for (int c = threadIdx.x; c < C; c += blockDim.x) {
    float sc = sb[c], bi = sb[C + c], s = 0.f;
    for (int n = 0; n < NPTS; ++n) {
      long long row = (long long)b * NPTS + n;
      s += fmaxf(Yf[row * C + c] * sc + bi, 0.f) * mask[row];
    }
    pooled[b * C + c] = s / cnt;
  }
}

// ---- final: extract first 2 of 16 padded cols, add bias ----
__global__ void out_extract(const float* __restrict__ fc2, const float* __restrict__ ob,
                            float* __restrict__ out) {
  int i = blockIdx.x * blockDim.x + threadIdx.x;
  if (i >= BATCH * 2) return;
  int b = i >> 1, c = i & 1;
  out[i] = fc2[b * 16 + c] + ob[c];
}

// ==========================================================================
extern "C" void kernel_launch(void* const* d_in, const int* in_sizes, int n_in,
                              void* d_out, int out_size, void* d_ws, size_t ws_size,
                              hipStream_t stream) {
  if (n_in < 48) return;
  auto f = [&](int i) { return (const float*)d_in[i]; };
  // pytree-flatten order (dict keys sorted; 'W' < lowercase):
  const float* points   = f(0);
  const float* features = f(1);
  const float* bnfts_g  = f(3);
  const float* bnfts_b  = f(4);
  struct ConvP { const float* W[3]; const float* g[3]; const float* b[3];
                 const float* scW; const float* scg; const float* scb; };
  ConvP cv[3];
  int ii = 5;
  for (int i = 0; i < 3; ++i) {
    for (int l = 0; l < 3; ++l) cv[i].W[l] = f(ii++);
    for (int l = 0; l < 3; ++l) { cv[i].g[l] = f(ii++); cv[i].b[l] = f(ii++); }
    cv[i].scW = f(ii++); cv[i].scg = f(ii++); cv[i].scb = f(ii++);
  }
  const float* fc_W = f(ii++); const float* fc_b = f(ii++);
  const float* fus_W = f(ii++); const float* fus_g = f(ii++); const float* fus_b = f(ii++);
  const float* out_W = f(ii++); const float* out_b = f(ii++);

  // ---------------- workspace carve ----------------
  char* w = (char*)d_ws;
  auto alloc = [&](size_t nfloats) { float* p = (float*)w; w += nfloats * sizeof(float); return p; };
  const long long ROWS = (long long)BATCH * NPTS * KNN; // 524288
  const long long RBN  = (long long)BATCH * NPTS;       // 32768
  float* bufA   = alloc((size_t)ROWS * 256);
  float* bufB   = alloc((size_t)ROWS * 256);
  float* fused  = alloc((size_t)RBN * 448);
  float* fts0   = alloc((size_t)RBN * 32);   // 7 real cols, zero-padded to 32
  float* SCb    = alloc((size_t)RBN * 256);
  float* pts0   = alloc((size_t)RBN * 2);
  float* mask   = alloc(RBN);
  float* counts = alloc(BATCH);
  int*   idx    = (int*)w; w += (size_t)RBN * KNN * sizeof(int);
  float* sums   = alloc(1024);
  float* sb     = alloc(1024);
  float* sb2    = alloc(1024);
  float* pooled = alloc((size_t)BATCH * 384);
  float* fc1    = alloc((size_t)BATCH * 256);
  float* fc2    = alloc((size_t)BATCH * 16); // padded final output
  _Float16* wpool = (_Float16*)w; w += (size_t)1024 * 1024 * sizeof(_Float16);
  size_t wp_off = 0;

  // pack weights (f32 row-major (Nc,K)) -> f16 fragment stream, padded
  auto packW = [&](const float* W, int K, int Kp, int Nc, int Ncp) {
    _Float16* dst = wpool + wp_off;
    int nK = Kp >> 5, nTiles = Ncp >> 4;
    int tot = nTiles * nK * 32;
    wp_off += (size_t)tot * 16;
    pack_w<<<(tot + 255) / 256, 256, 0, stream>>>(W, K, Nc, nK, nTiles, dst);
    return (const _Float16*)dst;
  };

  auto gemm = [&](const float* A, int Kp, const _Float16* Bp, const float* bias,
                  float* Cc, int ldc, long long M, int Ncp, int relu) {
    int mTiles = (int)(M >> 5);                // 32-row tiles
    int nTiles = Ncp >> 4;
    dim3 grid((unsigned)mTiles, (unsigned)((nTiles + 7) / 8));
    wmma_gemm_kernel<<<grid, 256, 0, stream>>>(A, Kp, Bp, bias, Cc, ldc, Kp, nTiles, relu);
  };
  auto stats = [&](const float* X, int ld, long long M, int C,
                   const float* g, const float* b, float* sbout) {
    zero_f32<<<(2 * C + 255) / 256, 256, 0, stream>>>(sums, 2 * C);
    bn_stats<<<1024, 256, 0, stream>>>(X, ld, M, C, sums);
    bn_finalize<<<(C + 127) / 128, 128, 0, stream>>>(sums, g, b, C, 1.f / (float)M, sbout);
  };
  auto apply = [&](float* X, int ld, long long M, int C, const float* sbp,
                   int relu, const float* mk) {
    long long tt = M * (long long)C;
    bn_apply<<<(unsigned)((tt + 255) / 256), 256, 0, stream>>>(X, ld, M, C, sbp, relu, mk);
  };

  // ---------------- pack all weights once ----------------
  const _Float16 *pW[3][3], *pSC[3];
  const int Cin_[3]  = {7, 64, 128};
  const int Cout_[3] = {64, 128, 256};
  const int Kp0_[3]  = {32, 128, 256};   // padded 2*Cin
  for (int i = 0; i < 3; ++i) {
    pW[i][0] = packW(cv[i].W[0], 2 * Cin_[i], Kp0_[i], Cout_[i], Cout_[i]);
    pW[i][1] = packW(cv[i].W[1], Cout_[i], Cout_[i], Cout_[i], Cout_[i]);
    pW[i][2] = packW(cv[i].W[2], Cout_[i], Cout_[i], Cout_[i], Cout_[i]);
    pSC[i]   = packW(cv[i].scW, Cin_[i], (i == 0) ? 32 : Cin_[i], Cout_[i], Cout_[i]);
  }
  const _Float16* pFus = packW(fus_W, 448, 448, 384, 384);
  const _Float16* pFc  = packW(fc_W, 384, 384, 256, 256);
  const _Float16* pOut = packW(out_W, 256, 256, 2, 16);

  // ---------------- preamble: mask, counts, BN(features) ----------------
  zero_f32<<<1, 256, 0, stream>>>(counts, BATCH);
  mask_kernel<<<(unsigned)((RBN + 255) / 256), 256, 0, stream>>>(
      features, points, mask, counts, fts0, pts0);
  stats(fts0, 32, RBN, 7, bnfts_g, bnfts_b, sb);
  apply(fts0, 32, RBN, 7, sb, /*relu*/0, mask);

  // ---------------- three EdgeConv blocks ----------------
  const int coff_[3] = {0, 64, 192};
  for (int blk = 0; blk < 3; ++blk) {
    int Cin = Cin_[blk], Co = Cout_[blk], Kp0 = Kp0_[blk];
    const float* ftsIn = (blk == 0) ? fts0 : (fused + coff_[blk - 1]);
    int ldIn           = (blk == 0) ? 32 : 448;
    const float* crd   = (blk == 0) ? pts0 : ftsIn;
    int ldCo           = (blk == 0) ? 2 : 448;
    int D              = (blk == 0) ? 2 : Cin;

    knn_kernel<<<BATCH, NPTS, (size_t)NPTS * D * sizeof(float), stream>>>(crd, ldCo, D, mask, idx);
    edge_feat<<<(unsigned)((ROWS + 255) / 256), 256, 0, stream>>>(ftsIn, ldIn, Cin, Kp0, idx, bufA);

    gemm(bufA, Kp0, pW[blk][0], nullptr, bufB, Co, ROWS, Co, 0);
    stats(bufB, Co, ROWS, Co, cv[blk].g[0], cv[blk].b[0], sb);
    apply(bufB, Co, ROWS, Co, sb, 1, nullptr);

    gemm(bufB, Co, pW[blk][1], nullptr, bufA, Co, ROWS, Co, 0);
    stats(bufA, Co, ROWS, Co, cv[blk].g[1], cv[blk].b[1], sb);
    apply(bufA, Co, ROWS, Co, sb, 1, nullptr);

    gemm(bufA, Co, pW[blk][2], nullptr, bufB, Co, ROWS, Co, 0);
    stats(bufB, Co, ROWS, Co, cv[blk].g[2], cv[blk].b[2], sb);   // BN applied in combine

    gemm(ftsIn, ldIn, pSC[blk], nullptr, SCb, Co, RBN, Co, 0);
    stats(SCb, Co, RBN, Co, cv[blk].scg, cv[blk].scb, sb2);

    combine_kernel<<<(unsigned)((RBN * Co + 255) / 256), 256, 0, stream>>>(
        bufB, Co, sb, SCb, sb2, mask, fused, coff_[blk]);
  }

  // ---------------- fusion conv + pool + FC head ----------------
  gemm(fused, 448, pFus, nullptr, bufB, 384, RBN, 384, 0);
  stats(bufB, 384, RBN, 384, fus_g, fus_b, sb);
  pool_kernel<<<BATCH, 256, 0, stream>>>(bufB, sb, mask, counts, pooled, 384);

  gemm(pooled, 384, pFc, fc_b, fc1, 256, BATCH, 256, 1);
  gemm(fc1, 256, pOut, nullptr, fc2, 16, BATCH, 16, 0);
  out_extract<<<2, 256, 0, stream>>>(fc2, out_b, (float*)d_out);

  (void)in_sizes; (void)out_size; (void)ws_size;
}
